// QuantizedLinear_74448963109154
// MI455X (gfx1250) — matrix operand order repaired and checked by reading, not compiled, not run on hardware
//
#include <hip/hip_runtime.h>

// ---- CDNA5 vector types ----
typedef __attribute__((ext_vector_type(16))) _Float16     v16h;
typedef __attribute__((ext_vector_type(8)))  _Float16     v8h;
typedef __attribute__((ext_vector_type(4)))  _Float16     v4h;
typedef __attribute__((ext_vector_type(2)))  _Float16     v2h;
typedef __attribute__((ext_vector_type(8)))  float        v8f;
typedef __attribute__((ext_vector_type(4)))  float        v4f;
typedef __attribute__((ext_vector_type(4)))  unsigned int v4u;

#define BM 128
#define BN 128
#define BK 64
#define LDK 72            // padded LDS row (f16 elems): 144B -> conflict-friendly
#define GROUP_SIZE 128

struct ARegs { v4f v[8]; };                          // 32 f32 of x per thread/step
struct BRegs { v4u q; _Float16 sch; _Float16 bth; }; // 32 int4 + f16 scale terms

__device__ __forceinline__ void load_a(ARegs& ar, const float* __restrict__ x,
                                       int m0, int K, int k0, int tid) {
#pragma unroll
  for (int j = 0; j < 8; ++j) {
    int i = tid + j * 256;          // 2048 float4 chunks: 128 rows x 16 chunks
    int r = i >> 4;
    int c = i & 15;
    ar.v[j] = *(const v4f*)(x + (size_t)(m0 + r) * K + k0 + c * 4);
  }
}

__device__ __forceinline__ void store_a(const ARegs& ar, _Float16* As, int tid) {
#pragma unroll
  for (int j = 0; j < 8; ++j) {
    int i = tid + j * 256;
    int r = i >> 4;
    int c = i & 15;
    v4f v = ar.v[j];
    v4h h;
#pragma unroll
    for (int e = 0; e < 4; ++e) h[e] = (_Float16)v[e];
    *(v4h*)(As + r * LDK + c * 4) = h;        // 8B ds_store, aligned
  }
}

__device__ __forceinline__ void load_b(BRegs& br, const int* __restrict__ qw,
                                       const float* __restrict__ scale,
                                       const float* __restrict__ zp,
                                       int n0, int Kq, int NG, int k0, int tid) {
  int n = tid >> 1;                 // 2 threads per N row
  int half = tid & 1;               // 32 K-values (4 int32) each
  const int* p = qw + (size_t)(n0 + n) * Kq + (k0 >> 3) + half * 4;
  br.q = *(const v4u*)p;
  int g = k0 / GROUP_SIZE;          // whole BK=64 chunk lies in one group
  float sc = scale[(size_t)(n0 + n) * NG + g];
  float z  = zp[(size_t)(n0 + n) * NG + g];
  br.sch = (_Float16)sc;
  br.bth = (_Float16)(-sc * z);     // w = sc*q - sc*zp
  __builtin_prefetch(p + 16, 0, 0); // global_prefetch_b8: qweight 2 K-steps ahead
}

// int4 -> f16 via magic mantissa: (0x6400 | q) == 1024.0 + q exactly.
// w = (h - 1024) * sc + (-sc*zp), packed f16 (v_pk_add_f16 + v_pk_fma_f16).
__device__ __forceinline__ void store_b(const BRegs& br, _Float16* Bs, int tid) {
  int n = tid >> 1;
  int half = tid & 1;
  v2h sc2; sc2[0] = br.sch; sc2[1] = br.sch;
  v2h bt2; bt2[0] = br.bth; bt2[1] = br.bth;
  v2h k1024; k1024[0] = (_Float16)1024.0f; k1024[1] = (_Float16)1024.0f;
#pragma unroll
  for (int j = 0; j < 4; ++j) {
    unsigned int qq = br.q[j];
    v8h h;
#pragma unroll
    for (int p = 0; p < 4; ++p) {
      unsigned int nib = qq >> (8 * p);    // nibble pair (2p, 2p+1) in low byte
      unsigned int pk = (nib & 0xFu) | ((nib & 0xF0u) << 12) | 0x64006400u;
      v2h hv = __builtin_bit_cast(v2h, pk);
      v2h w = (hv - k1024) * sc2 + bt2;    // exact sub, then pk_fma
      h[2 * p]     = w[0];
      h[2 * p + 1] = w[1];
    }
    *(v8h*)(Bs + n * LDK + half * 32 + j * 8) = h;   // 16B ds_store
  }
}

__device__ __forceinline__ v16h frag16(const _Float16* p, const _Float16* q) {
  v8h a = *(const v8h*)p;           // ds_load_b128
  v8h b = *(const v8h*)q;           // ds_load_b128
  v16h r;
#pragma unroll
  for (int i = 0; i < 8; ++i) { r[i] = a[i]; r[i + 8] = b[i]; }
  return r;
}

__global__ __launch_bounds__(256)
void qlinear_wmma_f16(const float* __restrict__ x,
                      const int* __restrict__ qweight,
                      const float* __restrict__ scale,
                      const float* __restrict__ zp,
                      const float* __restrict__ bias,
                      float* __restrict__ y,
                      int M, int N, int K)
{
  __shared__ __align__(16) _Float16 As[2][BM * LDK];  // 36 KB
  __shared__ __align__(16) _Float16 Bs[2][BN * LDK];  // 36 KB

  const int tid = threadIdx.x;
  const int m0 = blockIdx.y * BM;
  const int n0 = blockIdx.x * BN;
  const int Kq = K >> 3;            // int32 per weight row
  const int NG = K / GROUP_SIZE;    // groups per row

  // wave grid: 4 (M) x 2 (N); each wave: 32x64 = 2x4 16x16 accum tiles
  const int wave = tid >> 5;
  const int lane = tid & 31;
  const int wm = (wave & 3) * 32;
  const int wn = (wave >> 2) * 64;
  const int lh = lane & 15;
  const int ls = lane >> 4;

  v8f acc[2][4];
  {
    v8f zero;
#pragma unroll
    for (int i = 0; i < 8; ++i) zero[i] = 0.0f;
#pragma unroll
    for (int i = 0; i < 2; ++i)
#pragma unroll
      for (int j = 0; j < 4; ++j) acc[i][j] = zero;
  }

  ARegs ar; BRegs br;
  load_a(ar, x, m0, K, 0, tid);
  load_b(br, qweight, scale, zp, n0, Kq, NG, 0, tid);

  const int nsteps = K / BK;
  for (int ks = 0; ks < nsteps; ++ks) {
    const int buf = ks & 1;
    store_a(ar, As[buf], tid);      // f32 -> f16 convert + LDS stage
    store_b(br, Bs[buf], tid);      // int4 magic-dequant -> f16 LDS tile
    if (ks + 1 < nsteps) {          // issue next-step global loads early
      load_a(ar, x, m0, K, (ks + 1) * BK, tid);
      load_b(br, qweight, scale, zp, n0, Kq, NG, (ks + 1) * BK, tid);
    }
    __syncthreads();                // single barrier/iter: valid w/ dbl buffer

    const _Float16* Ab = As[buf];
    const _Float16* Bb = Bs[buf];
#pragma unroll
    for (int kf = 0; kf < BK; kf += 32) {
      // A frag (16x32 f16): lane lh = row; ls picks K {0-7,16-23} vs {8-15,24-31}
      int ra0 = (wm + lh) * LDK + kf + ls * 8;
      int ra1 = (wm + 16 + lh) * LDK + kf + ls * 8;
      v16h a0 = frag16(Ab + ra0, Ab + ra0 + 16);
      v16h a1 = frag16(Ab + ra1, Ab + ra1 + 16);
      // Preload all B frags (32x16 f16): lane lh = column; ls picks K {0-7,16-23} vs {8-15,24-31}
      v16h bf[4];
#pragma unroll
      for (int nf = 0; nf < 4; ++nf) {
        int rb = (wn + nf * 16 + lh) * LDK + kf + ls * 8;
        bf[nf] = frag16(Bb + rb, Bb + rb + 16);
      }
#pragma unroll
      for (int nf = 0; nf < 4; ++nf) {
        acc[0][nf] = __builtin_amdgcn_wmma_f32_16x16x32_f16(
            false, a0, false, bf[nf], (short)0, acc[0][nf], false, false);
        acc[1][nf] = __builtin_amdgcn_wmma_f32_16x16x32_f16(
            false, a1, false, bf[nf], (short)0, acc[1][nf], false, false);
      }
    }
  }
  __syncthreads();

  // epilogue: C layout — VGPR r: M=r (lanes 0-15), M=r+8 (lanes 16-31); N=lh
#pragma unroll
  for (int mi = 0; mi < 2; ++mi) {
#pragma unroll
    for (int nf = 0; nf < 4; ++nf) {
      int col = n0 + wn + nf * 16 + lh;
      float bv = bias[col];
      int rowb = m0 + wm + mi * 16 + ls * 8;
#pragma unroll
      for (int r = 0; r < 8; ++r) {
        y[(size_t)(rowb + r) * N + col] = acc[mi][nf][r] + bv;
      }
    }
  }
}

extern "C" void kernel_launch(void* const* d_in, const int* in_sizes, int n_in,
                              void* d_out, int out_size, void* d_ws, size_t ws_size,
                              hipStream_t stream) {
  const float* x    = (const float*)d_in[0];
  const int*   qw   = (const int*)d_in[1];
  const float* sc   = (const float*)d_in[2];
  const float* zp   = (const float*)d_in[3];
  const float* bias = (const float*)d_in[4];
  // d_in[5]=bits(4), d_in[6]=group_size(128): fixed by reference setup.

  int N = in_sizes[4];                                  // bias length
  long long K = ((long long)in_sizes[1] * 8) / N;       // qweight = N*K/8 int32
  int M = (int)((long long)in_sizes[0] / K);            // x = M*K

  dim3 grid((unsigned)(N / BN), (unsigned)(M / BM));
  qlinear_wmma_f16<<<grid, dim3(256), 0, stream>>>(
      x, qw, sc, zp, bias, (float*)d_out, M, N, (int)K);
}
